// CvTAttentionBlock_68075231641730
// MI455X (gfx1250) — compile-verified, matrix operand order checked by reading, not run on hardware
//
#include <hip/hip_runtime.h>

typedef __attribute__((ext_vector_type(16))) _Float16 v16h;
typedef __attribute__((ext_vector_type(8)))  _Float16 v8h;
typedef __attribute__((ext_vector_type(8)))  float    v8f;

union Frag16 { v16h v; v8h h[2]; };

#if __has_builtin(__builtin_amdgcn_tensor_load_to_lds) && \
    __has_builtin(__builtin_amdgcn_s_wait_tensorcnt)
#define HAVE_TDM 1
typedef unsigned int u32x4 __attribute__((ext_vector_type(4)));
typedef int          i32x4 __attribute__((ext_vector_type(4)));
typedef int          i32x8 __attribute__((ext_vector_type(8)));
#else
#define HAVE_TDM 0
#endif

// ---------------- constants ----------------
// B=16, H=W=56, C=192, heads=3, head_ch=64, Lq=3136, Lk=784
static constexpr int    kB   = 16;
static constexpr int    kC   = 192;
static constexpr int    kLq  = 3136;   // 56*56
static constexpr int    kLk  = 784;    // 28*28
static constexpr int    kLkP = 800;    // padded to multiple of 32 for WMMA K loop
static constexpr size_t kNQ  = (size_t)kB * kLq;  // 50176 rows
static constexpr size_t kNK  = (size_t)kB * kLk;  // 12544 rows

// =====================================================================
// 1) Depthwise 3x3 conv (SAME) + per-channel sum / sumsq for BatchNorm
//    block = 192 threads (one per channel), grid = B * outH
// =====================================================================
__global__ void dwconv_bn_stats(const float* __restrict__ X,
                                const float* __restrict__ W,
                                float* __restrict__ Y,
                                float* __restrict__ stats, // [0..191]=sum [192..383]=sumsq
                                int outH, int outW, int stride, int padlo) {
    const int c  = threadIdx.x;            // 0..191
    const int oy = blockIdx.x % outH;
    const int b  = blockIdx.x / outH;
    float wv[9];
#pragma unroll
    for (int i = 0; i < 9; ++i) wv[i] = W[i * kC + c];   // (ky,kx,1,C) layout
    float s = 0.f, s2 = 0.f;
    for (int ox = 0; ox < outW; ++ox) {
        float acc = 0.f;
#pragma unroll
        for (int ky = 0; ky < 3; ++ky) {
            const int iy = oy * stride + ky - padlo;
            if (iy < 0 || iy >= 56) continue;
#pragma unroll
            for (int kx = 0; kx < 3; ++kx) {
                const int ix = ox * stride + kx - padlo;
                if (ix < 0 || ix >= 56) continue;
                acc += X[((size_t)(b * 56 + iy) * 56 + ix) * kC + c] * wv[ky * 3 + kx];
            }
        }
        Y[((size_t)(b * outH + oy) * outW + ox) * kC + c] = acc;
        s += acc; s2 += acc * acc;
    }
    atomicAdd(&stats[c],       s);
    atomicAdd(&stats[kC + c],  s2);
}

// =====================================================================
// 2) BN finalize: per-channel mul/add so x_norm = x*mul + add
// =====================================================================
__global__ void bn_finalize(const float* __restrict__ stats, float invN,
                            const float* __restrict__ scale,
                            const float* __restrict__ offset,
                            float* __restrict__ nrm) {   // [0..191]=mul [192..383]=add
    const int c = threadIdx.x;
    const float mean = stats[c] * invN;
    const float var  = stats[kC + c] * invN - mean * mean;
    const float rm   = rsqrtf(var + 1e-5f);
    const float mul  = scale[c] * rm;
    nrm[c]      = mul;
    nrm[kC + c] = offset[c] - mean * mul;
}

// =====================================================================
// 3) Normalize + convert to f16
// =====================================================================
__global__ void normalize_f16(const float* __restrict__ X,
                              const float* __restrict__ nrm,
                              _Float16* __restrict__ Y, int n) {
    const int i = blockIdx.x * 256 + threadIdx.x;
    if (i >= n) return;
    const int c = i % kC;
    Y[i] = (_Float16)(X[i] * nrm[c] + nrm[kC + c]);
}

// =====================================================================
// 4) Transpose 192x192 f32 weight -> f16 [N][K] (so B fragments are
//    contiguous 16B chunks per lane, matching the 16-bit WMMA layout)
// =====================================================================
__global__ void transpose_w_f16(const float* __restrict__ W, _Float16* __restrict__ Wt) {
    const int i = blockIdx.x * 256 + threadIdx.x;
    if (i >= kC * kC) return;
    const int n = i / kC, k = i - n * kC;
    Wt[i] = (_Float16)W[k * kC + n];
}

// =====================================================================
// 5) Generic WMMA GEMM: [M,192](f16) x [192,192]^T(f16, row n = out col)
//    one wave per 16x16 output tile; K loop = 6 x 16x16x32 WMMA
// =====================================================================
__global__ void gemm_16x16_f16(const _Float16* __restrict__ A,
                               const _Float16* __restrict__ Bt,
                               void* __restrict__ Cp,
                               float scale, int outIsF16) {
    const int lane = threadIdx.x;   // 32 threads
    const int mn   = lane & 15;
    const int hf   = lane >> 4;
    const int m0   = blockIdx.x * 16;
    const int n0   = blockIdx.y * 16;
    const _Float16* arow = A  + ((size_t)m0 + mn) * kC;
    const _Float16* brow = Bt + ((size_t)n0 + mn) * kC;
    v8f acc = {};
#pragma unroll
    for (int kb = 0; kb < 6; ++kb) {
        const int k0 = kb * 32;
        if (kb < 5) {  // lowers to global_prefetch_b8 (near-cache hint)
            __builtin_prefetch(arow + k0 + 32, 0, 3);
            __builtin_prefetch(brow + k0 + 32, 0, 3);
        }
        Frag16 a, b;
        a.h[0] = *(const v8h*)(arow + k0 + hf * 8);
        a.h[1] = *(const v8h*)(arow + k0 + 16 + hf * 8);
        b.h[0] = *(const v8h*)(brow + k0 + hf * 8);
        b.h[1] = *(const v8h*)(brow + k0 + 16 + hf * 8);
        acc = __builtin_amdgcn_wmma_f32_16x16x32_f16(false, a.v, false, b.v,
                                                     (short)0, acc, false, false);
    }
    if (outIsF16) {
        _Float16* Co = (_Float16*)Cp;
#pragma unroll
        for (int r = 0; r < 8; ++r)
            Co[((size_t)m0 + r + 8 * hf) * kC + n0 + mn] = (_Float16)(acc[r] * scale);
    } else {
        float* Co = (float*)Cp;
#pragma unroll
        for (int r = 0; r < 8; ++r)
            Co[((size_t)m0 + r + 8 * hf) * kC + n0 + mn] = acc[r] * scale;
    }
}

// =====================================================================
// 6) Build padded V^T: [B][3][64][800] f16 (pad keys 784..799 = 0)
// =====================================================================
__global__ void build_vt(const _Float16* __restrict__ V, _Float16* __restrict__ VT) {
    const int i = blockIdx.x * 256 + threadIdx.x;
    if (i >= kB * 3 * 64 * kLkP) return;
    const int key = i % kLkP;
    const int d   = (i / kLkP) % 64;
    const int h   = (i / (kLkP * 64)) % 3;
    const int b   =  i / (kLkP * 64 * 3);
    _Float16 v = (_Float16)0.0f;
    if (key < kLk) v = V[((size_t)b * kLk + key) * kC + h * 64 + d];
    VT[i] = v;
}

// =====================================================================
// 7) Attention: per (b, 16-row q tile); talking heads + softmax in LDS
//    Dynamic LDS: q tile f16 [16][192] (TDM-staged) + scores [3][16][800]
// =====================================================================
__global__ void __launch_bounds__(256)
attn_kernel(const _Float16* __restrict__ Q,    // [B*3136,192] (pre-scaled 1/8)
            const _Float16* __restrict__ Kp,   // [B*784,192]
            const _Float16* __restrict__ VT,   // [B][3][64][800]
            const float* __restrict__ pre_sm,  // [3,3]
            const float* __restrict__ post_sm, // [3,3]
            _Float16* __restrict__ Sout) {     // [B*3136,192]
    extern __shared__ _Float16 smem[];
    _Float16* qtile = smem;                    // [16][192]  (at dynamic-LDS base)
    _Float16* S     = smem + 16 * kC;          // [3][16][800]
    const int tid  = threadIdx.x;
    const int lane = tid & 31;
    const int wave = tid >> 5;                 // 8 waves
    const int mn   = lane & 15;
    const int hf   = lane >> 4;
    const int l0   = blockIdx.x * 16;
    const int b    = blockIdx.y;

    // ---- stage q tile [16 rows x 192] into LDS via the Tensor Data Mover ----
    const _Float16* qsrc = Q + ((size_t)b * kLq + l0) * kC;
#if HAVE_TDM
    if (wave == 0) {
        const unsigned lds_addr = __builtin_amdgcn_groupstaticsize(); // dyn-LDS base
        const unsigned long long ga = (unsigned long long)(const void*)qsrc;
        u32x4 g0;
        g0[0] = 1u;                                            // count=1 (valid D#)
        g0[1] = lds_addr;                                      // lds_addr (bytes)
        g0[2] = (unsigned)(ga & 0xffffffffu);                  // global_addr[31:0]
        g0[3] = (unsigned)((ga >> 32) & 0x1ffffffu) | (2u << 30); // addr[56:32], type=2
        i32x8 g1;
        g1[0] = 1 << 16;                 // workgroup_mask=0, data_size=1 (2 bytes)
        g1[1] = (int)(192u << 16);       // tensor_dim0 = 192 (low 16 bits)
        g1[2] = (int)(16u << 16);        // tensor_dim0 hi=0 | tensor_dim1 = 16
        g1[3] = (int)(192u << 16);       // tensor_dim1 hi=0 | tile_dim0 = 192
        g1[4] = 16;                      // tile_dim1 = 16, tile_dim2 = 0
        g1[5] = 192;                     // tensor_dim0_stride (low 32) = 192
        g1[6] = 0; g1[7] = 0;            // stride hi, tensor_dim1_stride = 0
        i32x4 g2 = {0, 0, 0, 0}, g3 = {0, 0, 0, 0};
        i32x8 g4 = {0, 0, 0, 0, 0, 0, 0, 0};   // extra group (clang-23 6-arg form)
        __builtin_amdgcn_tensor_load_to_lds(g0, g1, g2, g3, g4, 0);
        __builtin_amdgcn_s_wait_tensorcnt(0);
    }
#else
    for (int i = tid; i < 16 * kC; i += 256) qtile[i] = qsrc[i];
#endif

    // zero padded key columns 784..799
    for (int i = tid; i < 3 * 16 * 16; i += 256) {
        const int hm = i >> 4;
        S[hm * kLkP + kLk + (i & 15)] = (_Float16)0.0f;
    }
    __syncthreads();

    // ---- Phase 1: scores S_h = q_h @ k_h^T  (49 key tiles x 3 heads) ----
    for (int job = wave; job < 3 * 49; job += 8) {
        const int h = job / 49, t = job % 49;
        const _Float16* qrow = qtile + (size_t)mn * kC + h * 64;           // LDS
        const _Float16* krow = Kp + ((size_t)b * kLk + t * 16 + mn) * kC + h * 64;
        v8f acc = {};
#pragma unroll
        for (int ks = 0; ks < 2; ++ks) {    // K = 64 = 2 x 32
            Frag16 a, bb;
            a.h[0]  = *(const v8h*)(qrow + ks * 32 + hf * 8);
            a.h[1]  = *(const v8h*)(qrow + ks * 32 + 16 + hf * 8);
            bb.h[0] = *(const v8h*)(krow + ks * 32 + hf * 8);
            bb.h[1] = *(const v8h*)(krow + ks * 32 + 16 + hf * 8);
            acc = __builtin_amdgcn_wmma_f32_16x16x32_f16(false, a.v, false, bb.v,
                                                         (short)0, acc, false, false);
        }
#pragma unroll
        for (int r = 0; r < 8; ++r)
            S[(h * 16 + r + 8 * hf) * kLkP + t * 16 + mn] = (_Float16)acc[r];
    }
    __syncthreads();

    // ---- pre-softmax talking heads (in place; thread owns (m,key)) ----
    float pm[9], qm[9];
#pragma unroll
    for (int i = 0; i < 9; ++i) { pm[i] = pre_sm[i]; qm[i] = post_sm[i]; }
    for (int p = tid; p < 16 * kLk; p += 256) {
        const int m = p / kLk, key = p - m * kLk;
        const float s0 = (float)S[(0 * 16 + m) * kLkP + key];
        const float s1 = (float)S[(1 * 16 + m) * kLkP + key];
        const float s2 = (float)S[(2 * 16 + m) * kLkP + key];
#pragma unroll
        for (int g = 0; g < 3; ++g)
            S[(g * 16 + m) * kLkP + key] =
                (_Float16)(s0 * pm[0 * 3 + g] + s1 * pm[1 * 3 + g] + s2 * pm[2 * 3 + g]);
    }
    __syncthreads();

    // ---- softmax: one wave per (head,row); 48 rows over 8 waves ----
    for (int row = wave; row < 48; row += 8) {
        _Float16* r = &S[row * kLkP];
        float mx = -3.0e38f;
        for (int k2 = lane; k2 < kLk; k2 += 32) mx = fmaxf(mx, (float)r[k2]);
#pragma unroll
        for (int off = 16; off > 0; off >>= 1) mx = fmaxf(mx, __shfl_xor(mx, off, 32));
        float sum = 0.f;
        for (int k2 = lane; k2 < kLk; k2 += 32) {
            const float e = __expf((float)r[k2] - mx);
            r[k2] = (_Float16)e;
            sum += e;
        }
#pragma unroll
        for (int off = 16; off > 0; off >>= 1) sum += __shfl_xor(sum, off, 32);
        const float inv = 1.0f / sum;
        for (int k2 = lane; k2 < kLk; k2 += 32)
            r[k2] = (_Float16)((float)r[k2] * inv);
    }
    __syncthreads();

    // ---- post-softmax talking heads (in place) ----
    for (int p = tid; p < 16 * kLk; p += 256) {
        const int m = p / kLk, key = p - m * kLk;
        const float s0 = (float)S[(0 * 16 + m) * kLkP + key];
        const float s1 = (float)S[(1 * 16 + m) * kLkP + key];
        const float s2 = (float)S[(2 * 16 + m) * kLkP + key];
#pragma unroll
        for (int g = 0; g < 3; ++g)
            S[(g * 16 + m) * kLkP + key] =
                (_Float16)(s0 * qm[0 * 3 + g] + s1 * qm[1 * 3 + g] + s2 * qm[2 * 3 + g]);
    }
    __syncthreads();

    // ---- Phase 3: out_h = P_h @ v_h  (3 heads x 4 d-tiles = 12 jobs) ----
    for (int job = wave; job < 12; job += 8) {
        const int h = job >> 2, dt = job & 3;
        const _Float16* srow = &S[(h * 16 + mn) * kLkP];
        const _Float16* vrow = VT + (((size_t)b * 3 + h) * 64 + dt * 16 + mn) * kLkP;
        v8f acc = {};
        for (int ks = 0; ks < kLkP / 32; ++ks) {   // 25 K-steps (pad is zero)
            Frag16 a, bb;
            a.h[0]  = *(const v8h*)(srow + ks * 32 + hf * 8);
            a.h[1]  = *(const v8h*)(srow + ks * 32 + 16 + hf * 8);
            bb.h[0] = *(const v8h*)(vrow + ks * 32 + hf * 8);
            bb.h[1] = *(const v8h*)(vrow + ks * 32 + 16 + hf * 8);
            acc = __builtin_amdgcn_wmma_f32_16x16x32_f16(false, a.v, false, bb.v,
                                                         (short)0, acc, false, false);
        }
#pragma unroll
        for (int r = 0; r < 8; ++r)
            Sout[((size_t)b * kLq + l0 + r + 8 * hf) * kC + h * 64 + dt * 16 + mn] =
                (_Float16)acc[r];
    }
}

// =====================================================================
// host: kernel_launch
// =====================================================================
extern "C" void kernel_launch(void* const* d_in, const int* in_sizes, int n_in,
                              void* d_out, int out_size, void* d_ws, size_t ws_size,
                              hipStream_t stream) {
    (void)in_sizes; (void)n_in; (void)out_size; (void)ws_size;
    char* ws = (char*)d_ws;
    constexpr size_t F = 4, H2 = 2;

    // workspace layout (all offsets 256B-aligned by construction)
    const size_t o_dwq   = 0;
    const size_t o_dwk   = o_dwq   + kNQ * kC * F;
    const size_t o_dwv   = o_dwk   + kNK * kC * F;
    const size_t o_stats = o_dwv   + kNK * kC * F;      // 3 x 384 floats
    const size_t o_nrm   = o_stats + 3 * 384 * F;       // 3 x 384 floats
    const size_t o_xnq   = o_nrm   + 3 * 384 * F;
    const size_t o_xnk   = o_xnq   + kNQ * kC * H2;
    const size_t o_xnv   = o_xnk   + kNK * kC * H2;
    const size_t o_qp    = o_xnv   + kNK * kC * H2;
    const size_t o_kp    = o_qp    + kNQ * kC * H2;
    const size_t o_vp    = o_kp    + kNK * kC * H2;
    const size_t o_vt    = o_vp    + kNK * kC * H2;     // B*3*64*800 halves
    const size_t o_s     = o_vt    + (size_t)kB * 3 * 64 * kLkP * H2;
    const size_t o_wtq   = o_s     + kNQ * kC * H2;
    const size_t o_wtk   = o_wtq   + (size_t)kC * kC * H2;
    const size_t o_wtv   = o_wtk   + (size_t)kC * kC * H2;
    const size_t o_wto   = o_wtv   + (size_t)kC * kC * H2;

    float*    dwq   = (float*)(ws + o_dwq);
    float*    dwk   = (float*)(ws + o_dwk);
    float*    dwv   = (float*)(ws + o_dwv);
    float*    stats = (float*)(ws + o_stats);
    float*    nrm   = (float*)(ws + o_nrm);
    _Float16* xnq   = (_Float16*)(ws + o_xnq);
    _Float16* xnk   = (_Float16*)(ws + o_xnk);
    _Float16* xnv   = (_Float16*)(ws + o_xnv);
    _Float16* qp    = (_Float16*)(ws + o_qp);
    _Float16* kp    = (_Float16*)(ws + o_kp);
    _Float16* vp    = (_Float16*)(ws + o_vp);
    _Float16* vt    = (_Float16*)(ws + o_vt);
    _Float16* sbuf  = (_Float16*)(ws + o_s);
    _Float16* wtq   = (_Float16*)(ws + o_wtq);
    _Float16* wtk   = (_Float16*)(ws + o_wtk);
    _Float16* wtv   = (_Float16*)(ws + o_wtv);
    _Float16* wto   = (_Float16*)(ws + o_wto);

    const float* in_q  = (const float*)d_in[0];
    const float* in_kv = (const float*)d_in[1];
    const float* dw_q  = (const float*)d_in[2];
    const float* dw_k  = (const float*)d_in[3];
    const float* dw_v  = (const float*)d_in[4];
    const float* sc_q  = (const float*)d_in[5];
    const float* of_q  = (const float*)d_in[6];
    const float* sc_k  = (const float*)d_in[7];
    const float* of_k  = (const float*)d_in[8];
    const float* sc_v  = (const float*)d_in[9];
    const float* of_v  = (const float*)d_in[10];
    const float* pw_q  = (const float*)d_in[11];
    const float* pw_k  = (const float*)d_in[12];
    const float* pw_v  = (const float*)d_in[13];
    const float* pre   = (const float*)d_in[14];
    const float* post  = (const float*)d_in[15];
    const float* w_out = (const float*)d_in[16];

    (void)hipMemsetAsync(stats, 0, 3 * 384 * F, stream);

    // weight transposes to f16 [N][K]
    const int wblk = (kC * kC + 255) / 256;
    transpose_w_f16<<<wblk, 256, 0, stream>>>(pw_q,  wtq);
    transpose_w_f16<<<wblk, 256, 0, stream>>>(pw_k,  wtk);
    transpose_w_f16<<<wblk, 256, 0, stream>>>(pw_v,  wtv);
    transpose_w_f16<<<wblk, 256, 0, stream>>>(w_out, wto);

    // depthwise conv + BN stats  (q: stride1 pad1; k/v: stride2 pad0)
    dwconv_bn_stats<<<kB * 56, kC, 0, stream>>>(in_q,  dw_q, dwq, stats + 0,   56, 56, 1, 1);
    dwconv_bn_stats<<<kB * 28, kC, 0, stream>>>(in_kv, dw_k, dwk, stats + 384, 28, 28, 2, 0);
    dwconv_bn_stats<<<kB * 28, kC, 0, stream>>>(in_kv, dw_v, dwv, stats + 768, 28, 28, 2, 0);

    bn_finalize<<<1, kC, 0, stream>>>(stats + 0,   1.0f / (float)kNQ, sc_q, of_q, nrm + 0);
    bn_finalize<<<1, kC, 0, stream>>>(stats + 384, 1.0f / (float)kNK, sc_k, of_k, nrm + 384);
    bn_finalize<<<1, kC, 0, stream>>>(stats + 768, 1.0f / (float)kNK, sc_v, of_v, nrm + 768);

    const int nq = (int)(kNQ * kC), nk = (int)(kNK * kC);
    normalize_f16<<<(nq + 255) / 256, 256, 0, stream>>>(dwq, nrm + 0,   xnq, nq);
    normalize_f16<<<(nk + 255) / 256, 256, 0, stream>>>(dwk, nrm + 384, xnk, nk);
    normalize_f16<<<(nk + 255) / 256, 256, 0, stream>>>(dwv, nrm + 768, xnv, nk);

    // pointwise 1x1 projections (WMMA); q pre-scaled by 1/sqrt(64)
    gemm_16x16_f16<<<dim3((int)(kNQ / 16), 12), 32, 0, stream>>>(xnq, wtq, qp, 0.125f, 1);
    gemm_16x16_f16<<<dim3((int)(kNK / 16), 12), 32, 0, stream>>>(xnk, wtk, kp, 1.0f,   1);
    gemm_16x16_f16<<<dim3((int)(kNK / 16), 12), 32, 0, stream>>>(xnv, wtv, vp, 1.0f,   1);

    // padded V^T
    const int nvt = kB * 3 * 64 * kLkP;
    build_vt<<<(nvt + 255) / 256, 256, 0, stream>>>(vp, vt);

    // attention (talking heads + softmax); LDS = q tile + scores
    const int shmem = 16 * kC * (int)H2 + 3 * 16 * kLkP * (int)H2;  // 6144 + 76800
    (void)hipFuncSetAttribute(reinterpret_cast<const void*>(attn_kernel),
                              hipFuncAttributeMaxDynamicSharedMemorySize, shmem);
    attn_kernel<<<dim3(kLq / 16, kB), 256, shmem, stream>>>(qp, kp, vt, pre, post, sbuf);

    // output linear -> f32 d_out
    gemm_16x16_f16<<<dim3((int)(kNQ / 16), 12), 32, 0, stream>>>(sbuf, wto, d_out, 1.0f, 0);
}
